// Model_19172734009798
// MI455X (gfx1250) — matrix operand order repaired and checked by reading, not compiled, not run on hardware
//
#include <hip/hip_runtime.h>
#include <cstdint>

// ---------------- model constants ----------------
constexpr int NB = 32;          // batch
constexpr int NT = 720;         // input time length
constexpr int NC = 862;         // channels
constexpr int NM = 4;           // covariates
constexpr int NLT = NC + NM;    // tokens = 866
constexpr int ND = 512;         // d_model
constexpr int NH = 8;           // heads
constexpr int NE = ND / NH;     // 64
constexpr int NFF = 2048;
constexpr int NLAYERS = 3;
constexpr int NPRED = 720;
constexpr int MTOK = NB * NLT;  // 27712 rows

typedef __attribute__((ext_vector_type(8)))  __bf16 v8bf;
typedef __attribute__((ext_vector_type(16))) __bf16 v16bf;
typedef __attribute__((ext_vector_type(8)))  float  v8f;
typedef unsigned int u32x4 __attribute__((ext_vector_type(4)));
typedef int i32x8 __attribute__((ext_vector_type(8)));
typedef int i32x4 __attribute__((ext_vector_type(4)));

__device__ __forceinline__ float sigmoidf_(float x) { return 1.0f / (1.0f + __expf(-x)); }
__device__ __forceinline__ float gelu_(float x)     { return 0.5f * x * (1.0f + erff(x * 0.70710678118654752f)); }

// Issue a TDM 2D tile load: mat[0..rows)[k0..k0+31] (bf16, row stride Krow elems)
// -> LDS at lds_addr, one 16B pad after every 64B row (LDS row pitch = 80B = 40 elems).
// OOB (cols >= dim0, rows >= dim1) reads return zero.
__device__ __forceinline__ void tdm_tile_load(const __bf16* mat, uint32_t lds_addr,
                                              uint32_t dim0, uint32_t dim1,
                                              int k0, int Krow) {
    const uint64_t ga = (uint64_t)(uintptr_t)(mat + k0);
    u32x4 g0;
    g0[0] = 1u;                                   // count=1, user descriptor
    g0[1] = lds_addr;                             // lds_addr (bytes)
    g0[2] = (uint32_t)ga;                         // global_addr[31:0]
    g0[3] = (uint32_t)((ga >> 32) & 0x1FFFFFFull) | (2u << 30);  // addr[56:32] | type=2
    i32x8 g1;
    g1[0] = (int)((1u << 16) | (1u << 20) | (3u << 22) | (3u << 25));
    //           data_size=2B | pad_enable | pad_interval=16DW | pad_amount=4DW
    g1[1] = (int)((dim0 & 0xFFFFu) << 16);        // tensor_dim0[15:0] @ bits63:48
    g1[2] = (int)((dim0 >> 16) | ((dim1 & 0xFFFFu) << 16));  // dim0[31:16] | dim1[15:0]
    g1[3] = (int)((dim1 >> 16) | (32u << 16));    // dim1[31:16] | tile_dim0=32
    g1[4] = (int)128u;                            // tile_dim1=128, tile_dim2=0
    g1[5] = (int)(uint32_t)Krow;                  // tensor_dim0_stride[31:0] (elements)
    g1[6] = 0;
    g1[7] = 0;
    i32x4 gz4; gz4[0] = 0; gz4[1] = 0; gz4[2] = 0; gz4[3] = 0;
    i32x8 gz8;
#pragma unroll
    for (int i = 0; i < 8; ++i) gz8[i] = 0;
    __builtin_amdgcn_tensor_load_to_lds(g0, g1, gz4, gz4, gz8, 0);
}

// ---------------- weight pack: f32 [K,N] -> bf16 [N,K] (transposed) ----------------
__global__ void pack_w_bf16_t(const float* __restrict__ src, __bf16* __restrict__ dst,
                              int Ksz, int Nsz) {
    int id = blockIdx.x * 256 + threadIdx.x;
    if (id < Ksz * Nsz) {
        int n = id / Ksz, k = id - n * Ksz;
        dst[(size_t)n * Ksz + k] = (__bf16)src[(size_t)k * Nsz + n];
    }
}

// ---------------- WMMA bf16 GEMM, A and B tiles both TDM-fed ----------------
// C[M,N] = A[M,K](bf16) * Bt[N,K](bf16)^T + bias ; optional f32 and/or bf16 outputs.
// Block tile 128x128, 256 threads = 8 waves (2x4); wave tile 64x32 (4x2 wmma frags).
// Both 128x32 tiles are DMA'd by the Tensor Data Mover into double-buffered,
// pad-interleaved LDS (wave 0 issues + TENSORcnt-waits; one barrier per K-step).
template <int EPI>  // 0 = bias, 1 = bias+GELU
__global__ __launch_bounds__(256) void gemm_bf16_wmma(
    const __bf16* __restrict__ A, const __bf16* __restrict__ Bt,
    const float* __restrict__ bias, float* __restrict__ Co, __bf16* __restrict__ Ch,
    int Msz, int Nsz, int Ksz) {
    __shared__ __align__(16) __bf16 AsT[2][128 * 40];   // 80B row pitch (TDM pad)
    __shared__ __align__(16) __bf16 BsT[2][128 * 40];

    const int tid  = threadIdx.x;
    const int lane = tid & 31;
    const int wave = tid >> 5;
    const int wm = wave >> 2, wn = wave & 3;     // 2 x 4 waves
    const int lr = lane & 15, hi = lane >> 4;
    const int mblk = blockIdx.y * 128, nblk = blockIdx.x * 128;

    const int ktiles = (Ksz + 31) / 32;
    const __bf16* Arow = A  + (size_t)mblk * Ksz;
    const __bf16* Brow = Bt + (size_t)nblk * Ksz;
    const uint32_t dimA1 = (uint32_t)(Msz - mblk);
    const uint32_t dimB1 = (uint32_t)(Nsz - nblk);

    v8f acc[4][2];
#pragma unroll
    for (int i = 0; i < 4; ++i)
#pragma unroll
        for (int j = 0; j < 2; ++j)
#pragma unroll
            for (int v = 0; v < 8; ++v) acc[i][j][v] = 0.0f;

    if (wave == 0) {
        tdm_tile_load(Arow, (uint32_t)(uintptr_t)(void*)&AsT[0][0], (uint32_t)Ksz, dimA1, 0, Ksz);
        tdm_tile_load(Brow, (uint32_t)(uintptr_t)(void*)&BsT[0][0], (uint32_t)Ksz, dimB1, 0, Ksz);
    }

    for (int kt = 0; kt < ktiles; ++kt) {
        if (wave == 0) __builtin_amdgcn_s_wait_tensorcnt(0);  // tiles kt landed in LDS
        __syncthreads();  // publishes tile kt; also proves buf kt+1 is no longer being read
        if (wave == 0 && kt + 1 < ktiles) {
            const int k1 = (kt + 1) * 32, bn = (kt + 1) & 1;
            tdm_tile_load(Arow, (uint32_t)(uintptr_t)(void*)&AsT[bn][0], (uint32_t)(Ksz - k1), dimA1, k1, Ksz);
            tdm_tile_load(Brow, (uint32_t)(uintptr_t)(void*)&BsT[bn][0], (uint32_t)(Ksz - k1), dimB1, k1, Ksz);
        }
        const __bf16* Asc = &AsT[kt & 1][0];
        const __bf16* Bsc = &BsT[kt & 1][0];

        // ---- B fragments: lane holds 16 consecutive K of column n (K-half by lane>=16) ----
        v16bf bfr[2];
#pragma unroll
        for (int nf = 0; nf < 2; ++nf) {
            const int n  = wn * 32 + nf * 16 + lr;
            const int kc = hi * 16;
            v8bf lo = *reinterpret_cast<const v8bf*>(&Bsc[n * 40 + kc]);
            v8bf h8 = *reinterpret_cast<const v8bf*>(&Bsc[n * 40 + kc + 8]);
            bfr[nf] = __builtin_shufflevector(lo, h8, 0, 1, 2, 3, 4, 5, 6, 7, 8, 9, 10, 11, 12, 13, 14, 15);
        }
        // ---- A fragments: lane<16 -> K 0..15 ; lane>=16 -> K 16..31 ----
#pragma unroll
        for (int mf = 0; mf < 4; ++mf) {
            const int r  = wm * 64 + mf * 16 + lr;
            const int kc = hi * 16;
            v8bf lo = *reinterpret_cast<const v8bf*>(&Asc[r * 40 + kc]);
            v8bf h8 = *reinterpret_cast<const v8bf*>(&Asc[r * 40 + kc + 8]);
            v16bf af = __builtin_shufflevector(lo, h8, 0, 1, 2, 3, 4, 5, 6, 7, 8, 9, 10, 11, 12, 13, 14, 15);
            acc[mf][0] = __builtin_amdgcn_wmma_f32_16x16x32_bf16(false, af, false, bfr[0], (short)0, acc[mf][0], false, false);
            acc[mf][1] = __builtin_amdgcn_wmma_f32_16x16x32_bf16(false, af, false, bfr[1], (short)0, acc[mf][1], false, false);
        }
    }

    // ---- epilogue: C/D layout: vgpr v, lane l -> m = v + 8*(l>=16), n = l%16 ----
#pragma unroll
    for (int mf = 0; mf < 4; ++mf)
#pragma unroll
        for (int nf = 0; nf < 2; ++nf) {
            const int n = nblk + wn * 32 + nf * 16 + lr;
#pragma unroll
            for (int v = 0; v < 8; ++v) {
                const int m = mblk + wm * 64 + mf * 16 + v + hi * 8;
                if (m < Msz && n < Nsz) {
                    float val = acc[mf][nf][v] + bias[n];
                    if (EPI == 1) val = gelu_(val);
                    if (Co) Co[(size_t)m * Nsz + n] = val;
                    if (Ch) Ch[(size_t)m * Nsz + n] = (__bf16)val;
                }
            }
        }
}

// ---------------- RevIN norm + token transpose: tok[b][c][t] (bf16) ----------------
__global__ __launch_bounds__(256) void revin_norm_kernel(
    const float* __restrict__ xe, const float* __restrict__ rw, const float* __restrict__ rb,
    __bf16* __restrict__ tokh, float* __restrict__ meanb, float* __restrict__ stdb) {
    const int b = blockIdx.y;
    const int cblk = blockIdx.x * 32;
    const int tc = threadIdx.x & 31, tg = threadIdx.x >> 5;
    const int c = cblk + tc;
    __shared__ float s_sum[8][32], s_sq[8][32], s_mean[32], s_std[32];
    float s = 0.f, q = 0.f;
    if (c < NC)
        for (int t = tg; t < NT; t += 8) {
            float v = xe[((size_t)b * NT + t) * NC + c];
            s += v; q += v * v;
        }
    s_sum[tg][tc] = s; s_sq[tg][tc] = q;
    __syncthreads();
    if (tg == 0) {
        float ss = 0.f, qq = 0.f;
#pragma unroll
        for (int g = 0; g < 8; ++g) { ss += s_sum[g][tc]; qq += s_sq[g][tc]; }
        float mean = ss / (float)NT;
        float var  = qq / (float)NT - mean * mean;
        float sd   = sqrtf(var + 1e-5f);
        s_mean[tc] = mean; s_std[tc] = sd;
        if (c < NC) { meanb[b * NC + c] = mean; stdb[b * NC + c] = sd; }
    }
    __syncthreads();
    for (int cc = 0; cc < 32; ++cc) {
        const int c2 = cblk + cc;
        if (c2 >= NC) break;
        const float mean = s_mean[cc], sd = s_std[cc];
        const float w = rw[c2], bb = rb[c2];
        for (int t = threadIdx.x; t < NT; t += 256)
            tokh[((size_t)b * NLT + c2) * NT + t] =
                (__bf16)((xe[((size_t)b * NT + t) * NC + c2] - mean) / sd * w + bb);
    }
}

__global__ void mark_tok_kernel(const float* __restrict__ xm, __bf16* __restrict__ tokh) {
    const int b = blockIdx.x;
    for (int i = threadIdx.x; i < NM * NT; i += 256) {
        int m = i / NT, t = i - m * NT;
        tokh[((size_t)b * NLT + NC + m) * NT + t] = (__bf16)xm[((size_t)b * NT + t) * NM + m];
    }
}

// ---------------- residual + LayerNorm over D=512; writes f32 (in place) + bf16 copy --------
__global__ __launch_bounds__(256) void add_ln_kernel(
    float* __restrict__ x, const float* __restrict__ y,
    const float* __restrict__ w, const float* __restrict__ bb, __bf16* __restrict__ xh) {
    const int row = blockIdx.x;
    const size_t base = (size_t)row * ND;
    const int tid = threadIdx.x;
    float v0 = x[base + tid], v1 = x[base + tid + 256];
    if (y) { v0 += y[base + tid]; v1 += y[base + tid + 256]; }
    __shared__ float rs[256], rq[256];
    rs[tid] = v0 + v1; rq[tid] = v0 * v0 + v1 * v1;
    __syncthreads();
    for (int off = 128; off > 0; off >>= 1) {
        if (tid < off) { rs[tid] += rs[tid + off]; rq[tid] += rq[tid + off]; }
        __syncthreads();
    }
    const float mean = rs[0] / (float)ND;
    const float var  = rq[0] / (float)ND - mean * mean;
    const float r    = rsqrtf(var + 1e-5f);
    const float o0 = (v0 - mean) * r * w[tid]       + bb[tid];
    const float o1 = (v1 - mean) * r * w[tid + 256] + bb[tid + 256];
    x[base + tid]       = o0;
    x[base + tid + 256] = o1;
    xh[base + tid]       = (__bf16)o0;
    xh[base + tid + 256] = (__bf16)o1;
}

// ---------------- Flowformer attention, one block per (b,h); bf16 output ----------------
__global__ __launch_bounds__(256) void flow_attn_kernel(
    const float* __restrict__ Q, const float* __restrict__ K,
    const float* __restrict__ V, __bf16* __restrict__ O) {
    const int bh = blockIdx.x;
    const int b = bh / NH, h = bh - b * NH;
    const int tid = threadIdx.x;
    const size_t base = (size_t)b * NLT * ND + (size_t)h * NE;
    const float eps = 1e-6f;

    __shared__ float s_ksum[NE], s_qsum[NE], s_kns[NE], s_qns[NE];
    __shared__ float s_nr[NLT], s_nc[NLT], s_ncref[NLT], s_nrs[NLT];
    __shared__ float s_red[256];
    __shared__ float s_kv[NE][NE + 1];
    __shared__ float s_kc[32][NE + 1];
    __shared__ float s_vc[32][NE + 1];

    // Pass A: ksum[e], qsum[e]
    {
        const int e = tid & 63, g = tid >> 6;
        float ks = 0.f, qs = 0.f;
        for (int l = g; l < NLT; l += 4) {
            ks += sigmoidf_(K[base + (size_t)l * ND + e]);
            qs += sigmoidf_(Q[base + (size_t)l * ND + e]);
        }
        s_red[tid] = ks; __syncthreads();
        if (g == 0) { float t = 0.f; for (int gg = 0; gg < 4; ++gg) t += s_red[e + (gg << 6)]; s_ksum[e] = t; }
        __syncthreads();
        s_red[tid] = qs; __syncthreads();
        if (g == 0) { float t = 0.f; for (int gg = 0; gg < 4; ++gg) t += s_red[e + (gg << 6)]; s_qsum[e] = t; }
        __syncthreads();
    }

    // Pass B: nr[l], nc[s]
    for (int l = tid; l < NLT; l += 256) {
        const float* qr = Q + base + (size_t)l * ND;
        const float* kr = K + base + (size_t)l * ND;
        float d1 = 0.f, d2 = 0.f;
        for (int e = 0; e < NE; ++e) {
            d1 += (sigmoidf_(qr[e]) + eps) * (s_ksum[e] + eps);
            d2 += (sigmoidf_(kr[e]) + eps) * (s_qsum[e] + eps);
        }
        s_nr[l] = 1.0f / d1;
        s_nc[l] = 1.0f / d2;
    }
    __syncthreads();

    // Pass C: kns[e] = sum_s k*nc ; qns[e] = sum_l q*nr
    {
        const int e = tid & 63, g = tid >> 6;
        float ks = 0.f, qs = 0.f;
        for (int l = g; l < NLT; l += 4) {
            ks += sigmoidf_(K[base + (size_t)l * ND + e]) * s_nc[l];
            qs += sigmoidf_(Q[base + (size_t)l * ND + e]) * s_nr[l];
        }
        s_red[tid] = ks; __syncthreads();
        if (g == 0) { float t = 0.f; for (int gg = 0; gg < 4; ++gg) t += s_red[e + (gg << 6)]; s_kns[e] = t; }
        __syncthreads();
        s_red[tid] = qs; __syncthreads();
        if (g == 0) { float t = 0.f; for (int gg = 0; gg < 4; ++gg) t += s_red[e + (gg << 6)]; s_qns[e] = t; }
        __syncthreads();
    }

    // Pass D: nrs[l] = nr*sigmoid(nr_ref), ncref[s] pre-softmax; then softmax * L
    float lmax = -3.4e38f;
    for (int l = tid; l < NLT; l += 256) {
        const float* qr = Q + base + (size_t)l * ND;
        const float* kr = K + base + (size_t)l * ND;
        float d1 = 0.f, d2 = 0.f;
        for (int e = 0; e < NE; ++e) {
            d1 += (sigmoidf_(qr[e]) + eps) * (s_kns[e] + eps);
            d2 += (sigmoidf_(kr[e]) + eps) * (s_qns[e] + eps);
        }
        s_nrs[l]   = s_nr[l] * sigmoidf_(d1 * ((float)NLT / (float)NLT));
        s_ncref[l] = d2;
        lmax = fmaxf(lmax, d2);
    }
    s_red[tid] = lmax; __syncthreads();
    for (int off = 128; off > 0; off >>= 1) {
        if (tid < off) s_red[tid] = fmaxf(s_red[tid], s_red[tid + off]);
        __syncthreads();
    }
    const float mx = s_red[0];
    __syncthreads();
    float lsum = 0.f;
    for (int l = tid; l < NLT; l += 256) {
        float ex = __expf(s_ncref[l] - mx);
        s_ncref[l] = ex;
        lsum += ex;
    }
    s_red[tid] = lsum; __syncthreads();
    for (int off = 128; off > 0; off >>= 1) {
        if (tid < off) s_red[tid] += s_red[tid + off];
        __syncthreads();
    }
    const float inv = (float)NLT / s_red[0];
    __syncthreads();
    for (int l = tid; l < NLT; l += 256) s_ncref[l] *= inv;
    __syncthreads();

    // Pass E: kv[e][d] = sum_s k[s][e] * v[s][d]*ncref[s]  (32-row LDS chunks)
    float accv[16];
#pragma unroll
    for (int j = 0; j < 16; ++j) accv[j] = 0.f;
    const int e_own = tid >> 2;
    const int dbase = (tid & 3) * 16;
    for (int s0 = 0; s0 < NLT; s0 += 32) {
        const int si = tid >> 3, c0 = (tid & 7) * 8;
        const int s = s0 + si;
#pragma unroll
        for (int j = 0; j < 8; ++j) {
            float kvl = 0.f, vvl = 0.f;
            if (s < NLT) {
                kvl = sigmoidf_(K[base + (size_t)s * ND + c0 + j]);
                vvl = V[base + (size_t)s * ND + c0 + j] * s_ncref[s];
            }
            s_kc[si][c0 + j] = kvl;
            s_vc[si][c0 + j] = vvl;
        }
        __syncthreads();
#pragma unroll 4
        for (int si2 = 0; si2 < 32; ++si2) {
            const float kk = s_kc[si2][e_own];
#pragma unroll
            for (int j = 0; j < 16; ++j) accv[j] += kk * s_vc[si2][dbase + j];
        }
        __syncthreads();
    }
#pragma unroll
    for (int j = 0; j < 16; ++j) s_kv[e_own][dbase + j] = accv[j];
    __syncthreads();

    // Pass F: out[l][d] = nrs[l] * sum_e sig(q[l][e]) * kv[e][d]  -> bf16
    const int lofs = tid >> 2;
    for (int l0 = 0; l0 < NLT; l0 += 64) {
        const int l = l0 + lofs;
        if (l < NLT) {
            const float* qr = Q + base + (size_t)l * ND;
            float o[16];
#pragma unroll
            for (int j = 0; j < 16; ++j) o[j] = 0.f;
            for (int e = 0; e < NE; ++e) {
                const float qe = sigmoidf_(qr[e]);
#pragma unroll
                for (int j = 0; j < 16; ++j) o[j] += qe * s_kv[e][dbase + j];
            }
            const float sc = s_nrs[l];
            __bf16* orow = O + base + (size_t)l * ND + dbase;
#pragma unroll
            for (int j = 0; j < 16; ++j) orow[j] = (__bf16)(o[j] * sc);
        }
    }
}

// ---------------- RevIN denorm + output transpose: out[b][p][c] ----------------
__global__ void denorm_out_kernel(const float* __restrict__ P,
                                  const float* __restrict__ rw, const float* __restrict__ rb,
                                  const float* __restrict__ meanb, const float* __restrict__ stdb,
                                  float* __restrict__ out) {
    size_t id = (size_t)blockIdx.x * 256 + threadIdx.x;
    const size_t tot = (size_t)NB * NPRED * NC;
    if (id >= tot) return;
    const int c = (int)(id % NC);
    const size_t r = id / NC;
    const int p = (int)(r % NPRED);
    const int b = (int)(r / NPRED);
    float v = P[((size_t)b * NLT + c) * NPRED + p];
    v = (v - rb[c]) / (rw[c] + 1e-10f);
    v = v * stdb[b * NC + c] + meanb[b * NC + c];
    out[id] = v;
}

// ---------------- host orchestration ----------------
extern "C" void kernel_launch(void* const* d_in, const int* in_sizes, int n_in,
                              void* d_out, int out_size, void* d_ws, size_t ws_size,
                              hipStream_t stream) {
    (void)in_sizes; (void)n_in; (void)out_size; (void)ws_size;
    const float* x_enc  = (const float*)d_in[0];
    const float* x_mark = (const float*)d_in[1];
    const float* revin_w = (const float*)d_in[4];
    const float* revin_b = (const float*)d_in[5];
    const float* emb_W = (const float*)d_in[6];
    const float* emb_b = (const float*)d_in[7];
    const float* Wq = (const float*)d_in[8];   const float* bq = (const float*)d_in[9];
    const float* Wk = (const float*)d_in[10];  const float* bk = (const float*)d_in[11];
    const float* Wv = (const float*)d_in[12];  const float* bv = (const float*)d_in[13];
    const float* Wo = (const float*)d_in[14];  const float* bo = (const float*)d_in[15];
    const float* ff1_W = (const float*)d_in[16]; const float* ff1_b = (const float*)d_in[17];
    const float* ff2_W = (const float*)d_in[18]; const float* ff2_b = (const float*)d_in[19];
    const float* ln1_w = (const float*)d_in[20]; const float* ln1_b = (const float*)d_in[21];
    const float* ln2_w = (const float*)d_in[22]; const float* ln2_b = (const float*)d_in[23];
    const float* lnf_w = (const float*)d_in[24]; const float* lnf_b = (const float*)d_in[25];
    const float* proj_W = (const float*)d_in[26]; const float* proj_b = (const float*)d_in[27];

    char* ws = (char*)d_ws;
    const size_t SZ_X  = (size_t)MTOK * ND * sizeof(float);   // 56.75 MB
    const size_t SZ_XH = (size_t)MTOK * ND * sizeof(__bf16);  // 28.4 MB
    size_t off = 0;
    auto arena = [&](size_t bytes) { size_t o = off; off += (bytes + 255) & ~(size_t)255; return o; };

    __bf16* tokh = (__bf16*)(ws + arena((size_t)MTOK * NT * 2));
    float*  Xb   = (float*)(ws + arena(SZ_X));
    __bf16* Xh   = (__bf16*)(ws + arena(SZ_XH));
    float*  Qb   = (float*)(ws + arena(SZ_X));   // Q..Ab region reused for FFH (bf16) and out_proj (f32)
    float*  Kb   = (float*)(ws + arena(SZ_X));
    float*  Vb   = (float*)(ws + arena(SZ_X));
    __bf16* Ab   = (__bf16*)(ws + arena(SZ_XH));
    float*  T1   = (float*)(ws + arena(SZ_X));
    float*  meanb = (float*)(ws + arena((size_t)NB * NC * 4));
    float*  stdb  = (float*)(ws + arena((size_t)NB * NC * 4));
    __bf16* FFHh  = (__bf16*)Qb;   // 27712 x 2048 bf16 = 113.5 MB <= 3*SZ_X + SZ_XH
    float*  out_proj = (float*)Qb; // 27712 x 720 f32 = 79.8 MB (after FFH dead)

    __bf16* w_emb = (__bf16*)(ws + arena((size_t)ND * NT * 2));
    __bf16* w_q[NLAYERS]; __bf16* w_k[NLAYERS]; __bf16* w_v[NLAYERS]; __bf16* w_o[NLAYERS];
    __bf16* w_f1[NLAYERS]; __bf16* w_f2[NLAYERS];
    for (int l = 0; l < NLAYERS; ++l) {
        w_q[l]  = (__bf16*)(ws + arena((size_t)ND * ND * 2));
        w_k[l]  = (__bf16*)(ws + arena((size_t)ND * ND * 2));
        w_v[l]  = (__bf16*)(ws + arena((size_t)ND * ND * 2));
        w_o[l]  = (__bf16*)(ws + arena((size_t)ND * ND * 2));
        w_f1[l] = (__bf16*)(ws + arena((size_t)NFF * ND * 2));
        w_f2[l] = (__bf16*)(ws + arena((size_t)ND * NFF * 2));
    }
    __bf16* w_proj = (__bf16*)(ws + arena((size_t)NPRED * ND * 2));

    auto pack = [&](const float* src, __bf16* dst, int Ksz, int Nsz) {
        int tot = Ksz * Nsz;
        pack_w_bf16_t<<<(tot + 255) / 256, 256, 0, stream>>>(src, dst, Ksz, Nsz);
    };
    auto gemm = [&](const __bf16* Ap, const __bf16* Btp, const float* biasp,
                    float* Cp, __bf16* Chp, int Ms, int Ns, int Ks, int epi) {
        dim3 g((Ns + 127) / 128, (Ms + 127) / 128);
        if (epi) gemm_bf16_wmma<1><<<g, 256, 0, stream>>>(Ap, Btp, biasp, Cp, Chp, Ms, Ns, Ks);
        else     gemm_bf16_wmma<0><<<g, 256, 0, stream>>>(Ap, Btp, biasp, Cp, Chp, Ms, Ns, Ks);
    };

    // weight packing (bf16, transposed to [N][K])
    pack(emb_W, w_emb, NT, ND);
    for (int l = 0; l < NLAYERS; ++l) {
        pack(Wq + (size_t)l * ND * ND, w_q[l], ND, ND);
        pack(Wk + (size_t)l * ND * ND, w_k[l], ND, ND);
        pack(Wv + (size_t)l * ND * ND, w_v[l], ND, ND);
        pack(Wo + (size_t)l * ND * ND, w_o[l], ND, ND);
        pack(ff1_W + (size_t)l * ND * NFF, w_f1[l], ND, NFF);
        pack(ff2_W + (size_t)l * NFF * ND, w_f2[l], NFF, ND);
    }
    pack(proj_W, w_proj, ND, NPRED);

    // RevIN + tokens (bf16)
    revin_norm_kernel<<<dim3((NC + 31) / 32, NB), 256, 0, stream>>>(
        x_enc, revin_w, revin_b, tokh, meanb, stdb);
    mark_tok_kernel<<<NB, 256, 0, stream>>>(x_mark, tokh);

    // embedding: X = tok @ emb_W + emb_b  (f32 for residual + bf16 for next GEMMs)
    gemm(tokh, w_emb, emb_b, Xb, Xh, MTOK, ND, NT, 0);

    for (int l = 0; l < NLAYERS; ++l) {
        gemm(Xh, w_q[l], bq + (size_t)l * ND, Qb, nullptr, MTOK, ND, ND, 0);
        gemm(Xh, w_k[l], bk + (size_t)l * ND, Kb, nullptr, MTOK, ND, ND, 0);
        gemm(Xh, w_v[l], bv + (size_t)l * ND, Vb, nullptr, MTOK, ND, ND, 0);
        flow_attn_kernel<<<NB * NH, 256, 0, stream>>>(Qb, Kb, Vb, Ab);
        gemm(Ab, w_o[l], bo + (size_t)l * ND, T1, nullptr, MTOK, ND, ND, 0);
        add_ln_kernel<<<MTOK, 256, 0, stream>>>(Xb, T1, ln1_w + (size_t)l * ND, ln1_b + (size_t)l * ND, Xh);
        gemm(Xh, w_f1[l], ff1_b + (size_t)l * NFF, nullptr, FFHh, MTOK, NFF, ND, 1);  // GELU, bf16 only
        gemm(FFHh, w_f2[l], ff2_b + (size_t)l * ND, T1, nullptr, MTOK, ND, NFF, 0);
        add_ln_kernel<<<MTOK, 256, 0, stream>>>(Xb, T1, ln2_w + (size_t)l * ND, ln2_b + (size_t)l * ND, Xh);
    }

    // final LN + projection + denorm
    add_ln_kernel<<<MTOK, 256, 0, stream>>>(Xb, nullptr, lnf_w, lnf_b, Xh);
    gemm(Xh, w_proj, proj_b, out_proj, nullptr, MTOK, NPRED, ND, 0);
    const size_t tot = (size_t)NB * NPRED * NC;
    denorm_out_kernel<<<(unsigned)((tot + 255) / 256), 256, 0, stream>>>(
        out_proj, revin_w, revin_b, meanb, stdb, (float*)d_out);
}
